// WaveletSNNTransformerModel_8108898255622
// MI455X (gfx1250) — matrix operand order repaired and checked by reading, not compiled, not run on hardware
//
#include <hip/hip_runtime.h>
#include <hip/hip_bf16.h>
#include <math.h>

typedef __attribute__((ext_vector_type(16))) _Float16 v16h;
typedef __attribute__((ext_vector_type(8)))  float    v8f;
typedef __attribute__((ext_vector_type(4)))  float    f4;
typedef __attribute__((ext_vector_type(4)))  _Float16 h4;

#define FLAG_TRANSB 1
#define FLAG_BETA   2
#define FLAG_RELU   4
#define FLAG_REMAP  8

constexpr int BM = 128, BN = 64, BK = 32;

// Generic batched GEMM: C[z] = act(alpha * A[z] @ B[z] + bias (+ C[z] if BETA))
// A: [M,K] fp32 row-major (lda). B: [K,N] (ldb) or [N,K] if TRANSB.
// fp32 -> fp16 conversion during LDS staging (b128 global loads, cvt_pk,
// b64 LDS stores, no branches: tail rows/cols are index-clamped, which is
// safe because WMMA rows/cols are independent and the epilogue guards all
// stores). Math is v_wmma_f32_16x16x32_f16 with fp32 accumulation.
template <bool TRANSB>
__global__ __launch_bounds__(256)
void gemm_wmma(const float* __restrict__ A, const float* __restrict__ B,
               const float* __restrict__ bias, float* __restrict__ C,
               int M, int N, int K, int lda, int ldb, int ldc,
               long long sA, long long sB, long long sC,
               int flags, float alpha, int rB, int rS)
{
    __shared__ _Float16 As[BM][BK];
    __shared__ _Float16 Bs[BK][BN + 8];

    const int tid  = threadIdx.x;
    const int lane = tid & 31, wid = tid >> 5;
    const int waveM = wid & 3, waveN = wid >> 2;   // 4 x 2 wave grid
    const int half = lane >> 4, l16 = lane & 15;
    const int z = blockIdx.z;
    const float* Ab = A + (long long)z * sA;
    const float* Bb = B + (long long)z * sB;
    float*       Cb = C + (long long)z * sC;
    const int m0 = blockIdx.y * BM;
    const int n0 = blockIdx.x * BN;

    // A staging map: thread -> (row, 16-col half-row); 4 x float4 per thread.
    // Row clamped in-bounds; tail-row garbage never escapes (epilogue guard).
    const int ar = tid >> 1;
    const int ac = (tid & 1) * 16;
    const int arClamp = min(m0 + ar, M - 1);
    const float* arowBase = Ab + (long long)arClamp * lda + ac;

    v8f zero = {};
    v8f acc[2][2];
    for (int i = 0; i < 2; ++i)
        for (int j = 0; j < 2; ++j) acc[i][j] = zero;

    for (int k0 = 0; k0 < K; k0 += BK) {
        // ---- Stage A tile (128x32): b128 loads, cvt_pk, b64 LDS stores ----
        #pragma unroll
        for (int u = 0; u < 4; ++u) {
            f4 v = *(const f4*)(arowBase + k0 + 4 * u);
            *(h4*)&As[ar][ac + 4 * u] = __builtin_convertvector(v, h4);
        }
        // ---- Stage B tile (32x64) ----
        if (!TRANSB) {
            // chunk over (k-row, n-float4): 512 chunks, 2 per thread
            #pragma unroll
            for (int u = 0; u < 2; ++u) {
                int chunk = tid + u * 256;
                int r  = chunk >> 4;               // k row 0..31
                int c4 = (chunk & 15) * 4;         // n offset
                int gn = min(n0 + c4, N - 4);      // clamp: tail never stored
                f4 v;
                __builtin_memcpy(&v, Bb + (long long)(k0 + r) * ldb + gn, 16);
                *(h4*)&Bs[r][c4] = __builtin_convertvector(v, h4);
            }
        } else {
            // B is [N,K]: load float4 along K (aligned), scatter-transpose
            #pragma unroll
            for (int u = 0; u < 2; ++u) {
                int chunk = tid + u * 256;
                int n  = chunk >> 3;               // n 0..63
                int k4 = (chunk & 7) * 4;          // k offset
                int gn = min(n0 + n, N - 1);       // clamp: tail never stored
                f4 v = *(const f4*)(Bb + (long long)gn * ldb + k0 + k4);
                h4 h = __builtin_convertvector(v, h4);
                #pragma unroll
                for (int e = 0; e < 4; ++e) Bs[k4 + e][n] = h[e];
            }
        }
        __syncthreads();

        // A fragment (ISA 7.12.2): lane -> M row; VGPR v: v<4 -> K=8*half+2v,
        // v>=4 -> K=16+8*half+2(v-4); packed pairs per VGPR.
        v16h af[2], bf[2];
        #pragma unroll
        for (int i = 0; i < 2; ++i) {
            int mrow = waveM * 32 + 16 * i + l16;
            #pragma unroll
            for (int e = 0; e < 16; ++e) {
                int v = e >> 1, s = e & 1;
                int kk = (v < 4) ? (half * 8 + 2 * v + s)
                                 : (16 + half * 8 + 2 * (v - 4) + s);
                af[i][e] = As[mrow][kk];
            }
        }
        // B fragment: lane -> N column l16; element e -> K in same order as A
        #pragma unroll
        for (int j = 0; j < 2; ++j) {
            int nb = waveN * 32 + 16 * j;
            #pragma unroll
            for (int e = 0; e < 16; ++e) {
                int kk = (e < 8) ? (half * 8 + e) : (16 + half * 8 + (e - 8));
                bf[j][e] = Bs[kk][nb + l16];
            }
        }
        #pragma unroll
        for (int i = 0; i < 2; ++i)
            #pragma unroll
            for (int j = 0; j < 2; ++j)
                acc[i][j] = __builtin_amdgcn_wmma_f32_16x16x32_f16(
                    false, af[i], false, bf[j], (short)0, acc[i][j], false, false);
        __syncthreads();
    }

    // C/D layout: acc element e -> M = e + 8*half, lane -> N = l16
    #pragma unroll
    for (int i = 0; i < 2; ++i) {
        #pragma unroll
        for (int j = 0; j < 2; ++j) {
            #pragma unroll
            for (int e = 0; e < 8; ++e) {
                int gm = m0 + waveM * 32 + 16 * i + 8 * half + e;
                int gn = n0 + waveN * 32 + 16 * j + l16;
                if (gm < M && gn < N) {
                    float val = acc[i][j][e] * alpha;
                    if (bias) val += bias[gn];
                    long long cidx;
                    if (flags & FLAG_REMAP) {
                        int orow = (gm % rB) * rS + (gm / rB);
                        cidx = (long long)orow * ldc + gn;
                    } else {
                        cidx = (long long)gm * ldc + gn;
                    }
                    if (flags & FLAG_BETA) val += Cb[cidx];
                    if (flags & FLAG_RELU) val = fmaxf(val, 0.f);
                    Cb[cidx] = val;
                }
            }
        }
    }
}

// x[s,b,d] = emb[src[b,s],d]*sqrt(D) + PE(s,d)
__global__ __launch_bounds__(256)
void embed_pe(const int* __restrict__ src, const float* __restrict__ emb,
              float* __restrict__ x, int Sd, int Bd, int Dd)
{
    long long idx = (long long)blockIdx.x * 256 + threadIdx.x;
    long long total = (long long)Sd * Bd * Dd;
    if (idx >= total) return;
    int d = (int)(idx % Dd);
    long long sb = idx / Dd;
    int b = (int)(sb % Bd);
    int s = (int)(sb / Bd);
    int tok = src[b * Sd + s];
    float e = emb[(long long)tok * Dd + d] * sqrtf((float)Dd);
    int i2 = d & ~1;
    float div = __expf(-(float)i2 * (logf(10000.f) / (float)Dd));
    float ang = (float)s * div;
    float pe = (d & 1) ? __cosf(ang) : __sinf(ang);
    x[idx] = e + pe;
}

// softmax over the first l+1 layer-mixing weights for every layer
__global__ void lw_softmax(const float* __restrict__ lw, float* __restrict__ out, int Lc)
{
    int l = blockIdx.x;
    if (threadIdx.x != 0) return;
    int n = l + 1;
    const float* row = lw + l * (Lc + 1);
    float mx = -1e30f;
    for (int j = 0; j < n; ++j) mx = fmaxf(mx, row[j]);
    float s = 0.f;
    for (int j = 0; j < n; ++j) s += expf(row[j] - mx);
    for (int j = 0; j < n; ++j) out[l * (Lc + 1) + j] = expf(row[j] - mx) / s;
}

// Per-row softmax over 256 keys, scaled by the layer-mix weight (uniform read)
__global__ __launch_bounds__(256)
void softmax_row(float* __restrict__ sc, const float* __restrict__ lwv)
{
    __shared__ float red[256];
    long long base = (long long)blockIdx.x * 256;
    int tid = threadIdx.x;
    float v = sc[base + tid];
    red[tid] = v; __syncthreads();
    for (int off = 128; off > 0; off >>= 1) {
        if (tid < off) red[tid] = fmaxf(red[tid], red[tid + off]);
        __syncthreads();
    }
    float mx = red[0]; __syncthreads();
    float e = expf(v - mx);
    red[tid] = e; __syncthreads();
    for (int off = 128; off > 0; off >>= 1) {
        if (tid < off) red[tid] += red[tid + off];
        __syncthreads();
    }
    float s = red[0];
    sc[base + tid] = (e / s) * (*lwv);
}

// Fused residual + LayerNorm over D=1024 (4 elems/thread); optional dup write
__global__ __launch_bounds__(256)
void ln_residual(float* __restrict__ x, const float* __restrict__ add,
                 const float* __restrict__ g, const float* __restrict__ be,
                 int Dd, float* __restrict__ dup)
{
    __shared__ float s1[256], s2[256];
    long long base = (long long)blockIdx.x * Dd;
    int tid = threadIdx.x;
    float v[4]; float a = 0.f, b2 = 0.f;
    #pragma unroll
    for (int t = 0; t < 4; ++t) {
        int col = tid + t * 256;
        float y = x[base + col] + add[base + col];
        v[t] = y; a += y; b2 += y * y;
    }
    s1[tid] = a; s2[tid] = b2; __syncthreads();
    for (int off = 128; off > 0; off >>= 1) {
        if (tid < off) { s1[tid] += s1[tid + off]; s2[tid] += s2[tid + off]; }
        __syncthreads();
    }
    float mean = s1[0] / (float)Dd;
    float var  = s2[0] / (float)Dd - mean * mean;
    float inv = rsqrtf(var + 1e-5f);
    #pragma unroll
    for (int t = 0; t < 4; ++t) {
        int col = tid + t * 256;
        float o = (v[t] - mean) * inv * g[col] + be[col];
        x[base + col] = o;
        if (dup) dup[base + col] = o;
    }
}

// Adaptive resonate-and-fire scan: sequential over S; one block per batch;
// thread d owns channel d; lateral inhibition needs sum_d(mem) every step.
__global__ __launch_bounds__(1024)
void snn_scan(const float* __restrict__ cur, float* __restrict__ sp,
              int Sd, int Bd, int Dd)
{
    __shared__ float red[1024];
    int b = blockIdx.x, d = threadIdx.x;
    float mem = 0.f, thr = 1.0f;
    for (int s = 0; s < Sd; ++s) {
        red[d] = mem; __syncthreads();
        for (int off = 512; off > 0; off >>= 1) {
            if (d < off) red[d] += red[d + off];
            __syncthreads();
        }
        float total = red[0]; __syncthreads();
        float c = cur[((long long)s * Bd + b) * Dd + d] - 0.1f * total;
        mem = 0.9f * mem + c;
        float spike = (mem >= thr) ? 1.f : 0.f;
        mem -= spike * thr;
        thr = 0.9f * thr + 0.1f * spike;
        sp[((long long)s * Bd + b) * Dd + d] = spike;
    }
}

extern "C" void kernel_launch(void* const* d_in, const int* in_sizes, int n_in,
                              void* d_out, int out_size, void* d_ws, size_t ws_size,
                              hipStream_t stream)
{
    (void)in_sizes; (void)n_in; (void)out_size; (void)ws_size;
    constexpr int Bc = 2, Sc = 256, Dc = 1024, Hc = 8, Lc = 16, FFc = 2048, Vc = 50257;
    constexpr int HDc = Dc / Hc;        // 128
    constexpr int SB  = Sc * Bc;        // 512 GEMM rows, [S,B,D] layout
    constexpr int BD  = Bc * Dc;        // 2048
    constexpr int BH  = Bc * Hc;        // 16 batched heads
    const float SCALE = 1.0f / sqrtf((float)HDc);

    const int*   src    = (const int*)  d_in[0];
    const float* emb    = (const float*)d_in[1];
    const float* Wq     = (const float*)d_in[2];
    const float* bq     = (const float*)d_in[3];
    const float* Wk     = (const float*)d_in[4];
    const float* bk     = (const float*)d_in[5];
    const float* Wv     = (const float*)d_in[6];
    const float* bv     = (const float*)d_in[7];
    const float* Wo     = (const float*)d_in[8];
    const float* bo     = (const float*)d_in[9];
    const float* layerw = (const float*)d_in[10];
    const float* Wsnn   = (const float*)d_in[11];
    const float* bsnn   = (const float*)d_in[12];
    const float* W1     = (const float*)d_in[13];
    const float* b1     = (const float*)d_in[14];
    const float* W2     = (const float*)d_in[15];
    const float* b2     = (const float*)d_in[16];
    const float* g1     = (const float*)d_in[17];
    const float* be1    = (const float*)d_in[18];
    const float* g2     = (const float*)d_in[19];
    const float* be2    = (const float*)d_in[20];
    const float* Wout   = (const float*)d_in[21];
    const float* bout   = (const float*)d_in[22];

    float* p = (float*)d_ws;
    auto alloc = [&](size_t nf) { float* r = p; p += nf; return r; };
    float* lw_sm = alloc(Lc * (Lc + 1));
    float* xbuf  = alloc((size_t)SB * Dc);
    float* lout[Lc];
    for (int l = 0; l < Lc; ++l) lout[l] = alloc((size_t)SB * Dc);
    float* qb   = alloc((size_t)SB * Dc);
    float* kb   = alloc((size_t)SB * Dc);
    float* vb   = alloc((size_t)SB * Dc);
    float* scb  = alloc((size_t)BH * Sc * Sc);
    float* comb = alloc((size_t)SB * Dc);
    float* ab   = alloc((size_t)SB * Dc);
    float* curb = alloc((size_t)SB * Dc);
    float* spb  = alloc((size_t)SB * Dc);
    float* ffh  = alloc((size_t)SB * FFc);
    float* ffo  = alloc((size_t)SB * Dc);

    auto gemm = [&](const float* A, const float* Bp, const float* bias, float* C,
                    int M, int N, int K, int lda, int ldb, int ldc,
                    long long sA, long long sB, long long sC, int batch,
                    int flags, float alpha) {
        dim3 grid((N + BN - 1) / BN, (M + BM - 1) / BM, batch);
        if (flags & FLAG_TRANSB)
            gemm_wmma<true><<<grid, dim3(256), 0, stream>>>(
                A, Bp, bias, C, M, N, K, lda, ldb, ldc, sA, sB, sC,
                flags, alpha, Bc, Sc);
        else
            gemm_wmma<false><<<grid, dim3(256), 0, stream>>>(
                A, Bp, bias, C, M, N, K, lda, ldb, ldc, sA, sB, sC,
                flags, alpha, Bc, Sc);
    };

    lw_softmax<<<Lc, 32, 0, stream>>>(layerw, lw_sm, Lc);
    embed_pe<<<(SB * Dc + 255) / 256, 256, 0, stream>>>(src, emb, xbuf, Sc, Bc, Dc);

    for (int l = 0; l < Lc; ++l) {
        const float* Wq_l = Wq + (size_t)l * Dc * Dc;   const float* bq_l = bq + (size_t)l * Dc;
        const float* Wk_l = Wk + (size_t)l * Dc * Dc;   const float* bk_l = bk + (size_t)l * Dc;
        const float* Wv_l = Wv + (size_t)l * Dc * Dc;   const float* bv_l = bv + (size_t)l * Dc;
        const float* Wo_l = Wo + (size_t)l * Dc * Dc;   const float* bo_l = bo + (size_t)l * Dc;
        const float* Ws_l = Wsnn + (size_t)l * Dc * Dc; const float* bs_l = bsnn + (size_t)l * Dc;
        const float* W1_l = W1 + (size_t)l * Dc * FFc;  const float* b1_l = b1 + (size_t)l * FFc;
        const float* W2_l = W2 + (size_t)l * FFc * Dc;  const float* b2_l = b2 + (size_t)l * Dc;

        // q = x @ Wq + bq
        gemm(xbuf, Wq_l, bq_l, qb, SB, Dc, Dc, Dc, Dc, Dc, 0, 0, 0, 1, 0, 1.f);

        int nsrc = l + 1;   // srcs = [x, out_0, ..., out_{l-1}]
        for (int n = 0; n < nsrc; ++n) {
            const float* sp_ = (n == 0) ? xbuf : lout[n - 1];
            gemm(sp_, Wk_l, bk_l, kb, SB, Dc, Dc, Dc, Dc, Dc, 0, 0, 0, 1, 0, 1.f);
            gemm(sp_, Wv_l, bv_l, vb, SB, Dc, Dc, Dc, Dc, Dc, 0, 0, 0, 1, 0, 1.f);
            // scores[z] = SCALE * q[z] @ k[z]^T   (head z offset = z*HD)
            gemm(qb, kb, nullptr, scb, Sc, Sc, HDc,
                 BD, BD, Sc, HDc, HDc, (long long)Sc * Sc, BH,
                 FLAG_TRANSB, SCALE);
            softmax_row<<<BH * Sc, 256, 0, stream>>>(scb, lw_sm + l * (Lc + 1) + n);
            // comb (+)= attn[z] @ v[z]  accumulated over sources, [S,B,D] out
            gemm(scb, vb, nullptr, comb, Sc, HDc, Sc,
                 Sc, BD, BD, (long long)Sc * Sc, HDc, HDc, BH,
                 (n > 0) ? FLAG_BETA : 0, 1.f);
        }
        gemm(comb, Wo_l, bo_l, ab, SB, Dc, Dc, Dc, Dc, Dc, 0, 0, 0, 1, 0, 1.f);
        ln_residual<<<SB, 256, 0, stream>>>(xbuf, ab, g1 + (size_t)l * Dc,
                                            be1 + (size_t)l * Dc, Dc, nullptr);

        gemm(xbuf, Ws_l, bs_l, curb, SB, Dc, Dc, Dc, Dc, Dc, 0, 0, 0, 1, 0, 1.f);
        snn_scan<<<Bc, 1024, 0, stream>>>(curb, spb, Sc, Bc, Dc);

        gemm(spb, W1_l, b1_l, ffh, SB, FFc, Dc, Dc, FFc, FFc, 0, 0, 0, 1, FLAG_RELU, 1.f);
        gemm(ffh, W2_l, b2_l, ffo, SB, Dc, FFc, FFc, Dc, Dc, 0, 0, 0, 1, 0, 1.f);
        ln_residual<<<SB, 256, 0, stream>>>(xbuf, ffo, g2 + (size_t)l * Dc,
                                            be2 + (size_t)l * Dc, Dc, lout[l]);
    }

    // logits = x @ Wout + bout, written directly to d_out with [S,B]->[B,S] remap
    gemm(xbuf, Wout, bout, (float*)d_out, SB, Vc, Dc, Dc, Vc, Vc,
         0, 0, 0, 1, FLAG_REMAP, 1.f);
}
